// PaperCompliantMoE_13761075216635
// MI455X (gfx1250) — matrix operand order repaired and checked by reading, not compiled, not run on hardware
//
#include <hip/hip_runtime.h>

// ---------------------------------------------------------------------------
// MI455X / gfx1250 fused MoE: router + top-2 gathered routed experts +
// TDM-staged shared experts. bf16 WMMA (v_wmma_f32_16x16x32_bf16), f32 acc.
// ---------------------------------------------------------------------------

typedef __attribute__((ext_vector_type(16))) __bf16 bf16x16;
typedef __attribute__((ext_vector_type(8)))  __bf16 bf16x8;
typedef __attribute__((ext_vector_type(8)))  float  f32x8;
typedef __attribute__((ext_vector_type(4)))  unsigned int u32x4;
typedef __attribute__((ext_vector_type(4)))  int i32x4;
typedef __attribute__((ext_vector_type(8)))  int i32x8;

#define N_TOK   2048
#define DIM     1024
#define NEXP    8
#define MSH     2
#define MT      32      // token tile rows per block
#define ICHUNK  128     // intermediate-dim chunk staged in LDS
#define THREADS 256     // 8 waves of 32

// ---- fragment helpers (ISA 7.12.2 layouts, wave32) ------------------------

__device__ __forceinline__ bf16x16 frag_cat(bf16x8 lo, bf16x8 hi) {
  return __builtin_shufflevector(lo, hi, 0,1,2,3,4,5,6,7,8,9,10,11,12,13,14,15);
}

// A matrix 16x32 bf16 from LDS row-major [rows][ld]:
// lane l: row m = l&15, khalf = (l>>4)*8; elems 0..7 = K khalf..+7,
// elems 8..15 = K 16+khalf..+7 (two 16B ds loads).
__device__ __forceinline__ bf16x16 load_a_lds(const __bf16* base, int ld,
                                              int row0, int k0, int lane) {
  const int m  = row0 + (lane & 15);
  const int kh = (lane >> 4) * 8;
  const __bf16* p = base + (size_t)m * ld + k0 + kh;
  bf16x8 lo = *(const bf16x8*)(p);
  bf16x8 hi = *(const bf16x8*)(p + 16);
  return frag_cat(lo, hi);
}

// B matrix 32x16 bf16 from fp32 global, row-major W (row stride ldw):
// B[k,n] = W[(n0+n)*ldw + k0 + k]; lane l: n = l&15, khalf = (l>>4)*8; elems 0..7 = K khalf..+7,
// elems 8..15 = K 16+khalf..+7.
__device__ __forceinline__ bf16x16 load_b_global_f32(const float* __restrict__ W,
                                                     int ldw, int n0, int k0,
                                                     int lane) {
  const float* p = W + (size_t)(n0 + (lane & 15)) * ldw + k0 + (lane >> 4) * 8;
  bf16x16 b;
#pragma unroll
  for (int j = 0; j < 4; ++j) {
    float4 f = ((const float4*)p)[j + (j & 2)];
    b[4*j+0] = (__bf16)f.x;
    b[4*j+1] = (__bf16)f.y;
    b[4*j+2] = (__bf16)f.z;
    b[4*j+3] = (__bf16)f.w;
  }
  return b;
}

__device__ __forceinline__ f32x8 wmma_bf16(bf16x16 a, bf16x16 b, f32x8 c) {
  return __builtin_amdgcn_wmma_f32_16x16x32_bf16(false, a, false, b,
                                                 (short)0, c, false, false);
}

// ---------------------------------------------------------------------------
// Fused SwiGLU expert MLP tile kernel.
//  GATHER=true : routed experts; token rows come from per-expert lists built
//                by the dispatch pass (top-2 sparsity -> 4x fewer FLOPs).
//  GATHER=false: shared experts; contiguous rows, x tile staged via the
//                Tensor Data Mover (tensor_load_to_lds + s_wait_tensorcnt),
//                then converted fp32->bf16 in place in LDS.
// ---------------------------------------------------------------------------
template <bool GATHER>
__global__ __launch_bounds__(THREADS)
void fused_swiglu_mlp(const float* __restrict__ x,        // [N, D]
                      const float* __restrict__ Wg_all,   // [nE, I, D]
                      const float* __restrict__ Wu_all,   // [nE, I, D]
                      const float* __restrict__ Wd_all,   // [nE, D, I]
                      const float* __restrict__ scale,    // [N, nE]
                      const int*   __restrict__ cnt,      // [nE]   (GATHER)
                      const int*   __restrict__ lists,    // [nE,N] (GATHER)
                      float* __restrict__ out,            // [N, D]
                      int N, int D, int I, int nE) {
  // 128 KB buffer: fp32 staging for TDM, aliased as the bf16 x tile.
  __shared__ __align__(16) unsigned char smem[MT * DIM * 4];
  __shared__ __bf16 inter_s[MT * ICHUNK];   // 8 KB
  __shared__ float  sc[MT];
  __shared__ int    tok_s[MT];
  float*  stage = (float*)smem;
  __bf16* xs    = (__bf16*)smem;

  const int e    = blockIdx.y;
  const int tok0 = blockIdx.x * MT;
  const int tid  = threadIdx.x;
  const int lane = tid & 31;
  const int wave = tid >> 5;

  int count = N;
  if (GATHER) count = cnt[e];
  if (tok0 >= count) return;   // empty tile for this expert: uniform early exit

  const float* Wg = Wg_all + (size_t)e * I * D;
  const float* Wu = Wu_all + (size_t)e * I * D;
  const float* Wd = Wd_all + (size_t)e * D * I;

  if constexpr (GATHER) {
    // ---- gather token ids + per-token combine weights ----
    if (tid < MT) {
      const int r     = tok0 + tid;
      const int valid = (r < count);
      const int t     = valid ? lists[(size_t)e * N + r] : 0;
      tok_s[tid] = t;
      sc[tid]    = valid ? scale[(size_t)t * nE + e] : 0.0f;
    }
    __syncthreads();
    // gathered x rows -> bf16 LDS
    for (int idx = tid * 4; idx < MT * D; idx += THREADS * 4) {
      const int m = idx / D, c = idx % D;
      const int token = tok_s[m];
      float4 f = *(const float4*)(x + (size_t)token * D + c);
      xs[idx + 0] = (__bf16)f.x;
      xs[idx + 1] = (__bf16)f.y;
      xs[idx + 2] = (__bf16)f.z;
      xs[idx + 3] = (__bf16)f.w;
    }
    __syncthreads();
  } else {
    // ---- TDM: DMA the contiguous [MT x D] fp32 tile into LDS ----
    if (wave == 0) {
      const unsigned long long ga =
          (unsigned long long)(const void*)(x + (size_t)tok0 * D);
      const unsigned lds_off = (unsigned)(size_t)(void*)smem;
      u32x4 g0;
      g0[0] = 1u;                                   // count=1 (user descriptor)
      g0[1] = lds_off;                              // lds_addr
      g0[2] = (unsigned)(ga & 0xFFFFFFFFu);         // global_addr[31:0]
      g0[3] = (unsigned)((ga >> 32) & 0x1FFFFFFu)   // global_addr[56:32]
              | (2u << 30);                         // type=2 ("image")
      i32x8 g1;
      g1[0] = (int)(2u << 16);                      // data_size=4B
      g1[1] = (int)((unsigned)(D & 0xFFFF) << 16);  // tensor_dim0[15:0]
      g1[2] = (int)(((unsigned)D >> 16) | ((unsigned)(N & 0xFFFF) << 16));
      g1[3] = (int)(((unsigned)N >> 16) | ((unsigned)D << 16)); // tile_dim0=D
      g1[4] = (int)MT;                              // tile_dim1=MT, tile_dim2=0
      g1[5] = (int)D;                               // tensor_dim0_stride lo32
      g1[6] = 0;
      g1[7] = 0;
      i32x4 gz = {0, 0, 0, 0};
#if defined(__clang_major__) && (__clang_major__ >= 23)
      i32x8 gz8 = {0, 0, 0, 0, 0, 0, 0, 0};
      __builtin_amdgcn_tensor_load_to_lds(g0, g1, gz, gz, gz8, 0);
#else
      __builtin_amdgcn_tensor_load_to_lds(g0, g1, gz, gz, 0);
#endif
      __builtin_amdgcn_s_wait_tensorcnt(0);
    }
    __syncthreads();
    // in-place fp32 -> bf16 (ascending chunks: writes at 2i stay below
    // any not-yet-read chunk at 4i; regs + barrier cover the self-chunk)
    constexpr int CH  = 8192;
    constexpr int PER = CH / THREADS;   // 32 floats per thread per chunk
    for (int c0 = 0; c0 < MT * DIM; c0 += CH) {
      float v[PER];
#pragma unroll
      for (int j = 0; j < PER; ++j) v[j] = stage[c0 + j * THREADS + tid];
      __syncthreads();
#pragma unroll
      for (int j = 0; j < PER; ++j) xs[c0 + j * THREADS + tid] = (__bf16)v[j];
      __syncthreads();
    }
    if (tid < MT) sc[tid] = scale[(size_t)(tok0 + tid) * nE + e];
    __syncthreads();
  }

  // Output accumulators: [MT=32, D=1024] as 8 d-tiles x 2 row-tiles per wave.
  f32x8 acc[16];
#pragma unroll
  for (int t = 0; t < 16; ++t) acc[t] = (f32x8){0,0,0,0,0,0,0,0};

  for (int ic = 0; ic < I; ic += ICHUNK) {
    // ---- stage 1: this wave computes inter columns [wave*16, wave*16+16) ----
    const int i0 = ic + wave * 16;
    f32x8 g0 = {0,0,0,0,0,0,0,0}, g1a = {0,0,0,0,0,0,0,0};
    f32x8 u0 = {0,0,0,0,0,0,0,0}, u1 = {0,0,0,0,0,0,0,0};
    for (int kd = 0; kd < D; kd += 32) {
      bf16x16 a0 = load_a_lds(xs, D,  0, kd, lane);
      bf16x16 a1 = load_a_lds(xs, D, 16, kd, lane);
      bf16x16 bg = load_b_global_f32(Wg, D, i0, kd, lane);
      bf16x16 bu = load_b_global_f32(Wu, D, i0, kd, lane);
      if (kd + 32 < D) {  // global_prefetch_b8 for next K slab
        __builtin_prefetch(Wg + (size_t)(i0 + (lane & 15)) * D + kd + 32, 0, 0);
        __builtin_prefetch(Wu + (size_t)(i0 + (lane & 15)) * D + kd + 32, 0, 0);
      }
      g0  = wmma_bf16(a0, bg, g0);
      g1a = wmma_bf16(a1, bg, g1a);
      u0  = wmma_bf16(a0, bu, u0);
      u1  = wmma_bf16(a1, bu, u1);
    }
    // inter = g * silu(u); C/D elem r of lane l is [m=(l>>4)*8+r, n=l&15]
    {
      const int n    = lane & 15;
      const int mrow = (lane >> 4) * 8;
#pragma unroll
      for (int r = 0; r < 8; ++r) {
        float g = g0[r], u = u0[r];
        float v = g * (u / (1.0f + __expf(-u)));
        inter_s[(size_t)(mrow + r) * ICHUNK + wave * 16 + n] = (__bf16)v;
        g = g1a[r]; u = u1[r];
        v = g * (u / (1.0f + __expf(-u)));
        inter_s[(size_t)(16 + mrow + r) * ICHUNK + wave * 16 + n] = (__bf16)v;
      }
    }
    __syncthreads();

    // ---- stage 2: down-projection partial, wave owns d in [wave*128,+128) ----
#pragma unroll
    for (int ki = 0; ki < ICHUNK; ki += 32) {
      bf16x16 a0 = load_a_lds(inter_s, ICHUNK,  0, ki, lane);
      bf16x16 a1 = load_a_lds(inter_s, ICHUNK, 16, ki, lane);
#pragma unroll
      for (int t = 0; t < 8; ++t) {
        const int d0 = wave * 128 + t * 16;
        bf16x16 bd = load_b_global_f32(Wd, I, d0, ic + ki, lane);
        acc[2*t + 0] = wmma_bf16(a0, bd, acc[2*t + 0]);
        acc[2*t + 1] = wmma_bf16(a1, bd, acc[2*t + 1]);
      }
    }
    __syncthreads();
  }

  // ---- epilogue: scale by per-token weight, accumulate into out ----
  const int n     = lane & 15;
  const int mbase = (lane >> 4) * 8;
#pragma unroll
  for (int t = 0; t < 8; ++t) {
    const int d0 = wave * 128 + t * 16;
#pragma unroll
    for (int rt = 0; rt < 2; ++rt) {
#pragma unroll
      for (int r = 0; r < 8; ++r) {
        const int m = rt * 16 + mbase + r;
        const int token = GATHER ? tok_s[m] : (tok0 + m);
        const float v = acc[2*t + rt][r] * sc[m];
        atomicAdd(&out[(size_t)token * D + d0 + n], v);
      }
    }
  }
}

// ---------------------------------------------------------------------------
// Router: one wave per token. 8 expert logits + 2 shared sigmoid gates.
// ---------------------------------------------------------------------------
__global__ __launch_bounds__(THREADS)
void router_kernel(const float* __restrict__ x,       // [N, D]
                   const float* __restrict__ Wr,      // [E, D]
                   const float* __restrict__ Wsgate,  // [MSH, D]
                   float* __restrict__ combine,       // [N, E]
                   float* __restrict__ sgate,         // [N, MSH]
                   float* __restrict__ probsum,       // [E]
                   int N, int D) {
  const int lane = threadIdx.x & 31;
  const int wave = threadIdx.x >> 5;
  const int tok  = blockIdx.x * (THREADS / 32) + wave;
  if (tok >= N) return;

  float acc[NEXP + MSH];
#pragma unroll
  for (int e = 0; e < NEXP + MSH; ++e) acc[e] = 0.0f;

  for (int d = lane; d < D; d += 32) {
    const float xv = x[(size_t)tok * D + d];
#pragma unroll
    for (int e = 0; e < NEXP; ++e) acc[e] += xv * Wr[(size_t)e * D + d];
    acc[NEXP + 0] += xv * Wsgate[d];
    acc[NEXP + 1] += xv * Wsgate[(size_t)D + d];
  }
#pragma unroll
  for (int e = 0; e < NEXP + MSH; ++e) {
#pragma unroll
    for (int off = 16; off > 0; off >>= 1)
      acc[e] += __shfl_xor(acc[e], off, 32);
  }

  if (lane == 0) {
    float mx = acc[0];
#pragma unroll
    for (int e = 1; e < NEXP; ++e) mx = fmaxf(mx, acc[e]);
    float p[NEXP]; float sum = 0.0f;
#pragma unroll
    for (int e = 0; e < NEXP; ++e) { p[e] = __expf(acc[e] - mx); sum += p[e]; }
    const float inv = 1.0f / sum;
#pragma unroll
    for (int e = 0; e < NEXP; ++e) p[e] *= inv;

    int i1 = 0;
#pragma unroll
    for (int e = 1; e < NEXP; ++e) if (p[e] > p[i1]) i1 = e;
    int i2 = (i1 == 0) ? 1 : 0;
#pragma unroll
    for (int e = 0; e < NEXP; ++e) if (e != i1 && p[e] > p[i2]) i2 = e;
    const float wsum = p[i1] + p[i2];
#pragma unroll
    for (int e = 0; e < NEXP; ++e) {
      const float c = (e == i1) ? p[i1] / wsum : ((e == i2) ? p[i2] / wsum : 0.0f);
      combine[(size_t)tok * NEXP + e] = c;
      atomicAdd(&probsum[e], p[e]);
    }
    sgate[(size_t)tok * MSH + 0] = 1.0f / (1.0f + __expf(-acc[NEXP + 0]));
    sgate[(size_t)tok * MSH + 1] = 1.0f / (1.0f + __expf(-acc[NEXP + 1]));
  }
}

// Build per-expert token lists from the top-2 combine weights.
__global__ void dispatch_kernel(const float* __restrict__ combine, // [N, E]
                                int* __restrict__ cnt,             // [E]
                                int* __restrict__ lists,           // [E, N]
                                int N) {
  const int n = blockIdx.x * blockDim.x + threadIdx.x;
  if (n >= N) return;
#pragma unroll
  for (int e = 0; e < NEXP; ++e) {
    if (combine[(size_t)n * NEXP + e] > 0.0f) {
      const int pos = atomicAdd(&cnt[e], 1);
      lists[(size_t)e * N + pos] = n;
    }
  }
}

__global__ void aux_kernel(const float* __restrict__ probsum,
                           float* __restrict__ out_aux, int N) {
  float s = 0.0f;
#pragma unroll
  for (int e = 0; e < NEXP; ++e) {
    const float a = probsum[e] / (float)N;
    s += a * a;
  }
  *out_aux = (float)NEXP * s * 0.001f;
}

__global__ void zero_kernel(float* __restrict__ p, size_t n) {
  const size_t i = (size_t)blockIdx.x * blockDim.x + threadIdx.x;
  if (i < n) p[i] = 0.0f;
}

// ---------------------------------------------------------------------------
extern "C" void kernel_launch(void* const* d_in, const int* in_sizes, int n_in,
                              void* d_out, int out_size, void* d_ws, size_t ws_size,
                              hipStream_t stream) {
  const float* x      = (const float*)d_in[0];  // [B,S,D]
  const float* Wr     = (const float*)d_in[1];  // [E,D]
  const float* Weg    = (const float*)d_in[2];  // [E,I,D]
  const float* Weu    = (const float*)d_in[3];  // [E,I,D]
  const float* Wed    = (const float*)d_in[4];  // [E,D,I]
  const float* Wsg    = (const float*)d_in[5];  // [M,ISH,D]
  const float* Wsu    = (const float*)d_in[6];  // [M,ISH,D]
  const float* Wsd    = (const float*)d_in[7];  // [M,D,ISH]
  const float* Wsgate = (const float*)d_in[8];  // [M,1,D]
  float* out = (float*)d_out;                   // [N*D] routed+shared, [1] aux

  const int N = N_TOK, D = DIM, IEXP = 2048, ISH = 5632;

  float* ws      = (float*)d_ws;
  float* combine = ws;                              // N*E   floats
  float* sgate   = combine + (size_t)N * NEXP;      // N*MSH floats
  float* probsum = sgate + (size_t)N * MSH;         // E     floats
  int*   cnt     = (int*)(probsum + NEXP);          // E     ints
  int*   lists   = cnt + NEXP;                      // E*N   ints

  // zero output + prob/count accumulators (graph-capture safe kernels)
  {
    const size_t nz = (size_t)out_size;
    zero_kernel<<<(unsigned)((nz + 255) / 256), 256, 0, stream>>>(out, nz);
    zero_kernel<<<1, 32, 0, stream>>>(probsum, (size_t)(NEXP + NEXP));
  }

  router_kernel<<<N / (THREADS / 32), THREADS, 0, stream>>>(
      x, Wr, Wsgate, combine, sgate, probsum, N, D);
  dispatch_kernel<<<N / 256, 256, 0, stream>>>(combine, cnt, lists, N);
  aux_kernel<<<1, 1, 0, stream>>>(probsum, out + (size_t)N * D, N);

  dim3 grid_r(N / MT, NEXP);   // routed experts (top-2 gathered)
  fused_swiglu_mlp<true><<<grid_r, THREADS, 0, stream>>>(
      x, Weg, Weu, Wed, combine, cnt, lists, out, N, D, IEXP, NEXP);

  dim3 grid_s(N / MT, MSH);    // gated shared experts (TDM-staged x tile)
  fused_swiglu_mlp<false><<<grid_s, THREADS, 0, stream>>>(
      x, Wsg, Wsu, Wsd, sgate, nullptr, nullptr, out, N, D, ISH, MSH);
}
